// BEVLayerInjector_33225867002512
// MI455X (gfx1250) — matrix operand order repaired and checked by reading, not compiled, not run on hardware
//
#include <hip/hip_runtime.h>
#include <hip/hip_bf16.h>

// ---------------------------------------------------------------------------
// BEVLayerInjector for MI455X (gfx1250):
//   - all GEMMs via v_wmma_f32_16x16x32_f16 (wave32, f32 accumulate)
//   - shared B panels staged in LDS by the Tensor Data Mover
//     (tensor_load_to_lds + s_wait_tensorcnt), double buffered
//   - chunk width is a template constant -> branch-free unrolled WMMA loop
// ---------------------------------------------------------------------------

typedef _Float16 half_t;
typedef __attribute__((ext_vector_type(16))) _Float16 v16h;
typedef __attribute__((ext_vector_type(8)))  _Float16 v8h;
typedef __attribute__((ext_vector_type(8)))  float    v8f;
typedef __attribute__((ext_vector_type(4))) unsigned int u32x4;
typedef __attribute__((ext_vector_type(8))) int          i32x8;
typedef __attribute__((ext_vector_type(4))) int          i32x4;

#define Bx    4
#define Sx    4096
#define HIDx  3584
#define BEVx  2048
#define DIMx  512
#define NHx   8
#define HDx   64
#define HWx   1024
#define NVx   1024
#define M4x   (Bx * NVx)

// ---------------------------------------------------------------------------
// TDM: DMA a 2-D f16 tile (rows x width) from global (row stride ldw halfs)
// into LDS at lds_off, padding each row by 4 DWORDs (pad_interval = row size).
// D# layout per CDNA5 ISA.
// ---------------------------------------------------------------------------
__device__ static inline void tdm_load_tile_f16(unsigned lds_off, const half_t* gsrc,
                                                int width_halfs, int rows, int ldw_halfs)
{
    unsigned long long ga = (unsigned long long)(uintptr_t)gsrc;
    u32x4 g0 = { 1u,                                   // count=1 (valid), user mode
                 lds_off,                              // LDS byte address
                 (unsigned)(ga & 0xFFFFFFFFu),
                 (unsigned)((ga >> 32) & 0x01FFFFFFu) | 0x80000000u };  // type=2
    const unsigned rw      = (unsigned)(width_halfs >> 1);    // row DWORDs (32 or 64)
    const unsigned padint  = (rw == 64u) ? 5u : 4u;           // pad every 2<<c DWORDs
    const unsigned padamt  = 3u;                              // +4 DWORDs per row
    const unsigned td0 = 1u << 30, td1 = 1u << 30;            // generous OOB dims
    const unsigned long long st0 = (unsigned long long)(unsigned)ldw_halfs;
    i32x8 g1;
    g1[0] = (int)((1u << 16) | (1u << 20) | (padint << 22) | (padamt << 25)); // 2B elems, pad_enable
    g1[1] = (int)((td0 & 0xFFFFu) << 16);                                     // tensor_dim0 lo16
    g1[2] = (int)(((td0 >> 16) & 0xFFFFu) | ((td1 & 0xFFFFu) << 16));         // dim0 hi | dim1 lo
    g1[3] = (int)(((td1 >> 16) & 0xFFFFu) | (((unsigned)width_halfs) << 16)); // dim1 hi | tile_dim0
    g1[4] = (int)((unsigned)rows & 0xFFFFu);                                  // tile_dim1 | tile_dim2=0
    g1[5] = (int)(st0 & 0xFFFFFFFFu);                                         // dim0_stride lo32
    g1[6] = (int)((st0 >> 32) & 0xFFFFu);                                     // stride hi | dim1_stride=0
    g1[7] = 0;
    i32x4 z4 = {0, 0, 0, 0};
#if __clang_major__ >= 23
    i32x8 z8 = {0, 0, 0, 0, 0, 0, 0, 0};
    __builtin_amdgcn_tensor_load_to_lds(g0, g1, z4, z4, z8, 0);
#else
    __builtin_amdgcn_tensor_load_to_lds(g0, g1, z4, z4, 0);
#endif
}

// ---------------------------------------------------------------------------
// Batched WMMA GEMM:  C[m,n] = act( alpha*sum_k A[m,k]*W[n,k] + bias[n] + resid )
// A: f16 row-major [M,K] (lda), W: f16 row-major [N,K] (ldw).
// Block = 8 waves: 128(M) x 64(N); B panel (64 x CW) is TDM-staged in LDS,
// shared by all 8 waves; A rows stream from global per wave.
// CW (K-chunk in halfs) is compile-time: 128 for K>=128, 64 for K==64.
// ---------------------------------------------------------------------------
template <bool GELU, int CW>
__global__ __launch_bounds__(256) void gemm_wmma_f16(
    const half_t* __restrict__ A, const half_t* __restrict__ W,
    const float* __restrict__ bias, const float* __restrict__ resid,
    float* __restrict__ Cf, half_t* __restrict__ Ch,
    int K, int lda, int ldw, int ldc,
    long sAb, long sAh, long sWb, long sWh, long sCb, long sCh,
    int hdiv, float alpha)
{
    constexpr unsigned STRIDE = (unsigned)(CW * 2 + 16);  // LDS row stride (bytes)
    constexpr unsigned BUFSZ  = 64u * STRIDE;
    __shared__ unsigned char smem[2 * BUFSZ] __attribute__((aligned(64)));

    const int z  = blockIdx.z;
    const int zb = z / hdiv;
    const int zh = z - zb * hdiv;
    const half_t* Ab = A + (long)zb * sAb + (long)zh * sAh;
    const half_t* Wb = W + (long)zb * sWb + (long)zh * sWh;
    const long coff  = (long)zb * sCb + (long)zh * sCh;

    const int lane = threadIdx.x & 31;
    const int wave = threadIdx.x >> 5;
    const int m0   = (blockIdx.y * 8 + wave) * 16;
    const int n0   = blockIdx.x * 64;
    const int lrow = lane & 15;
    const int hi   = lane >> 4;

    const int nc = K / CW;
    const unsigned ldsbase = (unsigned)(uintptr_t)&smem[0];

    // A fragment source: lane holds row (m0+lrow); chunks hi*8+[0..7], 16+hi*8+[0..7]
    const half_t* ap = Ab + (long)(m0 + lrow) * lda + hi * 8;
    // B tile source: 64 W-rows starting at n0
    const half_t* wtile = Wb + (long)n0 * ldw;

    v8f acc0 = {}, acc1 = {}, acc2 = {}, acc3 = {};

    if (wave == 0)   // prologue: DMA chunk 0 into buffer 0
        tdm_load_tile_f16(ldsbase, wtile, CW, 64, ldw);

    for (int c = 0; c < nc; ++c) {
        if (wave == 0) __builtin_amdgcn_s_wait_tensorcnt(0);
        __syncthreads();                                  // tile c visible to all waves
        if (wave == 0 && c + 1 < nc)                      // overlap DMA of next chunk
            tdm_load_tile_f16(ldsbase + (unsigned)((c + 1) & 1) * BUFSZ,
                              wtile + (long)(c + 1) * CW, CW, 64, ldw);

        const unsigned char* bbuf = smem + (unsigned)(c & 1) * BUFSZ;
        const int kb = c * CW;
#pragma unroll
        for (int kk = 0; kk < CW; kk += 32) {
            v16h a;
            ((v8h*)&a)[0] = *(const v8h*)(ap + kb + kk);
            ((v8h*)&a)[1] = *(const v8h*)(ap + kb + kk + 16);
            v16h b0, b1, b2, b3;
            {
                const unsigned coloff = (unsigned)(kk + hi * 8) * 2u;
                const unsigned char* r0 = bbuf + (unsigned)(lrow)      * STRIDE + coloff;
                const unsigned char* r1 = bbuf + (unsigned)(lrow + 16) * STRIDE + coloff;
                const unsigned char* r2 = bbuf + (unsigned)(lrow + 32) * STRIDE + coloff;
                const unsigned char* r3 = bbuf + (unsigned)(lrow + 48) * STRIDE + coloff;
                ((v8h*)&b0)[0] = *(const v8h*)(r0);  ((v8h*)&b0)[1] = *(const v8h*)(r0 + 32);
                ((v8h*)&b1)[0] = *(const v8h*)(r1);  ((v8h*)&b1)[1] = *(const v8h*)(r1 + 32);
                ((v8h*)&b2)[0] = *(const v8h*)(r2);  ((v8h*)&b2)[1] = *(const v8h*)(r2 + 32);
                ((v8h*)&b3)[0] = *(const v8h*)(r3);  ((v8h*)&b3)[1] = *(const v8h*)(r3 + 32);
            }
            acc0 = __builtin_amdgcn_wmma_f32_16x16x32_f16(false, a, false, b0, (short)0, acc0, false, false);
            acc1 = __builtin_amdgcn_wmma_f32_16x16x32_f16(false, a, false, b1, (short)0, acc1, false, false);
            acc2 = __builtin_amdgcn_wmma_f32_16x16x32_f16(false, a, false, b2, (short)0, acc2, false, false);
            acc3 = __builtin_amdgcn_wmma_f32_16x16x32_f16(false, a, false, b3, (short)0, acc3, false, false);
        }
        __syncthreads();                                  // all reads of tile c done
    }

    v8f accs[4] = {acc0, acc1, acc2, acc3};
#pragma unroll
    for (int j = 0; j < 4; j++) {
        const int n  = n0 + j * 16 + lrow;
        const float bv = bias ? bias[n] : 0.0f;
#pragma unroll
        for (int r = 0; r < 8; r++) {
            const long idx = coff + (long)(m0 + hi * 8 + r) * ldc + n;
            float v = accs[j][r] * alpha + bv;
            if (resid) v += resid[idx];
            if (GELU)  v = 0.5f * v * (1.0f + erff(v * 0.70710678118654752f));
            if (Cf) Cf[idx] = v;
            if (Ch) Ch[idx] = (half_t)v;
        }
    }
}

// ---------------------------------------------------------------------------
// Helpers
// ---------------------------------------------------------------------------
__global__ __launch_bounds__(256) void f32_to_f16(const float* __restrict__ in,
                                                  half_t* __restrict__ out, int n)
{
    int i = blockIdx.x * 256 + threadIdx.x;
    if (i < n) out[i] = (half_t)in[i];
}

__global__ __launch_bounds__(256) void gather_f16(const float* __restrict__ hs,
                                                  const int* __restrict__ vis_idx,
                                                  half_t* __restrict__ out)
{
    const int row = blockIdx.x;          // 0 .. B*NV-1
    const int b   = row >> 10;           // NV = 1024
    const int idx = vis_idx[row];
    const float* src = hs + ((long)b * Sx + idx) * HIDx;
    half_t* dst = out + (long)row * HIDx;
    for (int c = threadIdx.x; c < HIDx; c += 256) dst[c] = (half_t)src[c];
}

// bev_feat [B, BEV, HW] -> Abev f16 [B, HW, BEV]
__global__ __launch_bounds__(256) void bev_transpose_f16(const float* __restrict__ in,
                                                         half_t* __restrict__ out)
{
    __shared__ float tile[32][33];
    const int b  = blockIdx.z;
    const int c0 = blockIdx.x * 32;
    const int p0 = blockIdx.y * 32;
    const int tx = threadIdx.x, ty = threadIdx.y;   // (32, 8)
#pragma unroll
    for (int i = 0; i < 4; i++)
        tile[ty + i * 8][tx] = in[((long)(b * BEVx + c0 + ty + i * 8)) * HWx + p0 + tx];
    __syncthreads();
#pragma unroll
    for (int i = 0; i < 4; i++)
        out[((long)(b * HWx + p0 + ty + i * 8)) * BEVx + c0 + tx] = (half_t)tile[tx][ty + i * 8];
}

// v f16 [B*NV, DIM] -> vT f16 [B, H, HD, NV]
__global__ __launch_bounds__(256) void v_transpose_f16(const half_t* __restrict__ v,
                                                       half_t* __restrict__ vT)
{
    __shared__ half_t tile[32][33];
    const int z = blockIdx.z;            // b*8 + h
    const int b = z >> 3, h = z & 7;
    const int m0 = blockIdx.x * 32;
    const int d0 = blockIdx.y * 32;
    const int tx = threadIdx.x, ty = threadIdx.y;
#pragma unroll
    for (int i = 0; i < 4; i++)
        tile[ty + i * 8][tx] = v[((long)(b * NVx + m0 + ty + i * 8)) * DIMx + h * HDx + d0 + tx];
    __syncthreads();
#pragma unroll
    for (int i = 0; i < 4; i++)
        vT[((long)(z * HDx + d0 + ty + i * 8)) * NVx + m0 + tx] = tile[tx][ty + i * 8];
}

__global__ __launch_bounds__(128) void softmax_rows_f16(half_t* __restrict__ s)
{
    __shared__ float red[128];
    const long base = (long)blockIdx.x * HWx;
    const int  t = threadIdx.x;
    float mx = -1e30f;
    for (int i = t; i < HWx; i += 128) mx = fmaxf(mx, (float)s[base + i]);
    red[t] = mx; __syncthreads();
    for (int o = 64; o > 0; o >>= 1) { if (t < o) red[t] = fmaxf(red[t], red[t + o]); __syncthreads(); }
    mx = red[0]; __syncthreads();
    float sum = 0.0f;
    for (int i = t; i < HWx; i += 128) sum += expf((float)s[base + i] - mx);
    red[t] = sum; __syncthreads();
    for (int o = 64; o > 0; o >>= 1) { if (t < o) red[t] += red[t + o]; __syncthreads(); }
    const float inv = 1.0f / red[0];
    for (int i = t; i < HWx; i += 128)
        s[base + i] = (half_t)(expf((float)s[base + i] - mx) * inv);
}

__global__ __launch_bounds__(256) void layernorm_rows(const float* __restrict__ x,
                                                      const float* __restrict__ g,
                                                      const float* __restrict__ bta,
                                                      half_t* __restrict__ out)
{
    __shared__ float rs[256], rs2[256];
    const long base = (long)blockIdx.x * DIMx;
    const int  t = threadIdx.x;
    float s = 0.0f, s2 = 0.0f;
    for (int i = t; i < DIMx; i += 256) { float v = x[base + i]; s += v; s2 += v * v; }
    rs[t] = s; rs2[t] = s2; __syncthreads();
    for (int o = 128; o > 0; o >>= 1) {
        if (t < o) { rs[t] += rs[t + o]; rs2[t] += rs2[t + o]; }
        __syncthreads();
    }
    const float mu   = rs[0] * (1.0f / DIMx);
    const float var  = rs2[0] * (1.0f / DIMx) - mu * mu;
    const float rstd = rsqrtf(var + 1e-5f);
    for (int i = t; i < DIMx; i += 256)
        out[base + i] = (half_t)((x[base + i] - mu) * rstd * g[i] + bta[i]);
}

__global__ __launch_bounds__(256) void scatter_add(const float* __restrict__ hs,
                                                   const int* __restrict__ vis_idx,
                                                   const float* __restrict__ y,
                                                   float* __restrict__ out)
{
    const int row = blockIdx.x;
    const int b   = row >> 10;
    const int idx = vis_idx[row];
    const long o  = ((long)b * Sx + idx) * (long)HIDx;
    const float* yr = y + (long)row * HIDx;
    for (int c = threadIdx.x; c < HIDx; c += 256) out[o + c] = hs[o + c] + yr[c];
}

// ---------------------------------------------------------------------------
// Host launch
// ---------------------------------------------------------------------------
extern "C" void kernel_launch(void* const* d_in, const int* in_sizes, int n_in,
                              void* d_out, int out_size, void* d_ws, size_t ws_size,
                              hipStream_t stream)
{
    (void)in_sizes; (void)n_in; (void)out_size; (void)ws_size;

    const float* hs   = (const float*)d_in[0];
    const float* bev  = (const float*)d_in[1];
    const int*   vidx = (const int*)  d_in[2];
    const float* w1a = (const float*)d_in[3],  * b1a = (const float*)d_in[4];
    const float* w1b = (const float*)d_in[5],  * b1b = (const float*)d_in[6];
    const float* w2a = (const float*)d_in[7],  * b2a = (const float*)d_in[8];
    const float* w2b = (const float*)d_in[9],  * b2b = (const float*)d_in[10];
    const float* wq  = (const float*)d_in[11], * bq  = (const float*)d_in[12];
    const float* wk  = (const float*)d_in[13], * bk  = (const float*)d_in[14];
    const float* wv  = (const float*)d_in[15], * bv  = (const float*)d_in[16];
    const float* wo  = (const float*)d_in[17], * bo  = (const float*)d_in[18];
    const float* lng = (const float*)d_in[19], * lnb = (const float*)d_in[20];
    const float* w3a = (const float*)d_in[21], * b3a = (const float*)d_in[22];
    const float* w3b = (const float*)d_in[23], * b3b = (const float*)d_in[24];
    float* out = (float*)d_out;

    char* p = (char*)d_ws;
    auto take = [&](size_t bytes) -> char* {
        char* r = p; p += (bytes + 255) & ~(size_t)255; return r;
    };
    half_t* w1a_h = (half_t*)take((size_t)DIMx * HIDx * 2);
    half_t* w1b_h = (half_t*)take((size_t)DIMx * DIMx * 2);
    half_t* w2a_h = (half_t*)take((size_t)DIMx * BEVx * 2);
    half_t* w2b_h = (half_t*)take((size_t)DIMx * DIMx * 2);
    half_t* wq_h  = (half_t*)take((size_t)DIMx * DIMx * 2);
    half_t* wk_h  = (half_t*)take((size_t)DIMx * DIMx * 2);
    half_t* wv_h  = (half_t*)take((size_t)DIMx * DIMx * 2);
    half_t* wo_h  = (half_t*)take((size_t)DIMx * DIMx * 2);
    half_t* w3a_h = (half_t*)take((size_t)DIMx * DIMx * 2);
    half_t* w3b_h = (half_t*)take((size_t)HIDx * DIMx * 2);
    half_t* Avis  = (half_t*)take((size_t)M4x * HIDx * 2);
    half_t* Abev  = (half_t*)take((size_t)M4x * BEVx * 2);
    half_t* h1    = (half_t*)take((size_t)M4x * DIMx * 2);
    float*  vh_f  = (float*) take((size_t)M4x * DIMx * 4);
    half_t* vh_h  = (half_t*)take((size_t)M4x * DIMx * 2);
    half_t* g1    = (half_t*)take((size_t)M4x * DIMx * 2);
    half_t* bh_h  = (half_t*)take((size_t)M4x * DIMx * 2);
    half_t* q_h   = (half_t*)take((size_t)M4x * DIMx * 2);
    half_t* k_h   = (half_t*)take((size_t)M4x * DIMx * 2);
    half_t* v_h   = (half_t*)take((size_t)M4x * DIMx * 2);
    half_t* s_h   = (half_t*)take((size_t)Bx * NHx * NVx * HWx * 2);
    half_t* vT_h  = (half_t*)take((size_t)Bx * NHx * HDx * NVx * 2);
    half_t* ao_h  = (half_t*)take((size_t)M4x * DIMx * 2);
    float*  x_f   = (float*) take((size_t)M4x * DIMx * 4);
    half_t* f_h   = (half_t*)take((size_t)M4x * DIMx * 2);
    half_t* h3    = (half_t*)take((size_t)M4x * DIMx * 2);
    float*  y_f   = (float*) take((size_t)M4x * HIDx * 4);

    auto cvt = [&](const float* src, half_t* dst, int n) {
        f32_to_f16<<<(n + 255) / 256, 256, 0, stream>>>(src, dst, n);
    };
    cvt(w1a, w1a_h, DIMx * HIDx);  cvt(w1b, w1b_h, DIMx * DIMx);
    cvt(w2a, w2a_h, DIMx * BEVx);  cvt(w2b, w2b_h, DIMx * DIMx);
    cvt(wq,  wq_h,  DIMx * DIMx);  cvt(wk,  wk_h,  DIMx * DIMx);
    cvt(wv,  wv_h,  DIMx * DIMx);  cvt(wo,  wo_h,  DIMx * DIMx);
    cvt(w3a, w3a_h, DIMx * DIMx);  cvt(w3b, w3b_h, HIDx * DIMx);

    gather_f16<<<M4x, 256, 0, stream>>>(hs, vidx, Avis);
    bev_transpose_f16<<<dim3(BEVx / 32, HWx / 32, Bx), dim3(32, 8), 0, stream>>>(bev, Abev);

    auto gemm = [&](bool gelu, const half_t* A, const half_t* W, const float* bias,
                    const float* resid, float* Cf, half_t* Ch,
                    int M, int N, int K, int lda, int ldw, int ldc,
                    long sAb, long sAh, long sWb, long sWh, long sCb, long sCh,
                    int batch, int hdiv, float alpha) {
        dim3 grid(N / 64, M / 128, batch);
        if (gelu)
            gemm_wmma_f16<true, 128><<<grid, 256, 0, stream>>>(A, W, bias, resid, Cf, Ch,
                K, lda, ldw, ldc, sAb, sAh, sWb, sWh, sCb, sCh, hdiv, alpha);
        else if (K >= 128)
            gemm_wmma_f16<false, 128><<<grid, 256, 0, stream>>>(A, W, bias, resid, Cf, Ch,
                K, lda, ldw, ldc, sAb, sAh, sWb, sWh, sCb, sCh, hdiv, alpha);
        else
            gemm_wmma_f16<false, 64><<<grid, 256, 0, stream>>>(A, W, bias, resid, Cf, Ch,
                K, lda, ldw, ldc, sAb, sAh, sWb, sWh, sCb, sCh, hdiv, alpha);
    };

    // vision MLP
    gemm(true,  Avis, w1a_h, b1a, nullptr, nullptr, h1,
         M4x, DIMx, HIDx, HIDx, HIDx, DIMx, 0,0,0,0,0,0, 1, 1, 1.0f);
    gemm(false, h1, w1b_h, b1b, nullptr, vh_f, vh_h,
         M4x, DIMx, DIMx, DIMx, DIMx, DIMx, 0,0,0,0,0,0, 1, 1, 1.0f);
    // bev MLP
    gemm(true,  Abev, w2a_h, b2a, nullptr, nullptr, g1,
         M4x, DIMx, BEVx, BEVx, BEVx, DIMx, 0,0,0,0,0,0, 1, 1, 1.0f);
    gemm(false, g1, w2b_h, b2b, nullptr, nullptr, bh_h,
         M4x, DIMx, DIMx, DIMx, DIMx, DIMx, 0,0,0,0,0,0, 1, 1, 1.0f);
    // projections
    gemm(false, vh_h, wq_h, bq, nullptr, nullptr, q_h,
         M4x, DIMx, DIMx, DIMx, DIMx, DIMx, 0,0,0,0,0,0, 1, 1, 1.0f);
    gemm(false, bh_h, wk_h, bk, nullptr, nullptr, k_h,
         M4x, DIMx, DIMx, DIMx, DIMx, DIMx, 0,0,0,0,0,0, 1, 1, 1.0f);
    gemm(false, bh_h, wv_h, bv, nullptr, nullptr, v_h,
         M4x, DIMx, DIMx, DIMx, DIMx, DIMx, 0,0,0,0,0,0, 1, 1, 1.0f);
    // scores: per (b,h)  S = (Q @ K^T)/8  -> f16 [B,H,NV,HW]   (K = HD = 64)
    gemm(false, q_h, k_h, nullptr, nullptr, nullptr, s_h,
         NVx, HWx, HDx, DIMx, DIMx, HWx,
         (long)NVx * DIMx, HDx, (long)HWx * DIMx, HDx,
         (long)NHx * NVx * HWx, (long)NVx * HWx, Bx * NHx, NHx, 0.125f);
    softmax_rows_f16<<<Bx * NHx * NVx, 128, 0, stream>>>(s_h);
    v_transpose_f16<<<dim3(NVx / 32, HDx / 32, Bx * NHx), dim3(32, 8), 0, stream>>>(v_h, vT_h);
    // attention output: per (b,h)  AO = P @ V
    gemm(false, s_h, vT_h, nullptr, nullptr, nullptr, ao_h,
         NVx, HDx, HWx, HWx, NVx, DIMx,
         (long)NHx * NVx * HWx, (long)NVx * HWx, (long)NHx * HDx * NVx, (long)HDx * NVx,
         (long)NVx * DIMx, HDx, Bx * NHx, NHx, 1.0f);
    // x = vh + (ao @ wo^T + bo)
    gemm(false, ao_h, wo_h, bo, vh_f, x_f, nullptr,
         M4x, DIMx, DIMx, DIMx, DIMx, DIMx, 0,0,0,0,0,0, 1, 1, 1.0f);
    layernorm_rows<<<M4x, 256, 0, stream>>>(x_f, lng, lnb, f_h);
    // output MLP
    gemm(true,  f_h, w3a_h, b3a, nullptr, nullptr, h3,
         M4x, DIMx, DIMx, DIMx, DIMx, DIMx, 0,0,0,0,0,0, 1, 1, 1.0f);
    gemm(false, h3, w3b_h, b3b, nullptr, y_f, nullptr,
         M4x, HIDx, DIMx, DIMx, DIMx, HIDx, 0,0,0,0,0,0, 1, 1, 1.0f);

    // out = hidden_states; out[(b, vis_idx)] = hidden + y
    hipMemcpyAsync(out, hs, (size_t)Bx * Sx * HIDx * sizeof(float),
                   hipMemcpyDeviceToDevice, stream);
    scatter_add<<<M4x, 256, 0, stream>>>(hs, vidx, y_f, out);
}
